// SympleAgent_21947282882856
// MI455X (gfx1250) — matrix operand order repaired and checked by reading, not compiled, not run on hardware
//
#include <hip/hip_runtime.h>
#include <hip/hip_bf16.h>

typedef _Float16 v16h __attribute__((ext_vector_type(16)));
typedef _Float16 h8   __attribute__((ext_vector_type(8)));
typedef float    v8f  __attribute__((ext_vector_type(8)));
typedef int      v4i  __attribute__((ext_vector_type(4)));

#ifndef __has_builtin
#define __has_builtin(x) 0
#endif

#if __has_builtin(__builtin_amdgcn_global_load_async_to_lds_b128)
#define HAVE_ASYNC 1
typedef __attribute__((address_space(1))) v4i GV4;   // global int4
typedef __attribute__((address_space(3))) v4i LV4;   // LDS int4
#else
#define HAVE_ASYNC 0
#endif

__device__ __forceinline__ float sigm(float x) { return 1.0f / (1.0f + __expf(-x)); }

// 16-byte global -> LDS copy; async DMA path on gfx1250 when available.
__device__ __forceinline__ void g2l_b128(const _Float16* g, _Float16* l) {
#if HAVE_ASYNC
  __builtin_amdgcn_global_load_async_to_lds_b128(
      (GV4*)(unsigned long long)(uintptr_t)g,
      (LV4*)(unsigned)(uintptr_t)l, 0, 0);
#else
  *(h8*)l = *(const h8*)g;
#endif
}

__device__ __forceinline__ void wait_async() {
#if HAVE_ASYNC
#if __has_builtin(__builtin_amdgcn_s_wait_asynccnt)
  __builtin_amdgcn_s_wait_asynccnt(0);
#else
  asm volatile("s_wait_asynccnt 0" ::: "memory");
#endif
#endif
}

// ---------------------------------------------------------------------------
// Pack the level weights into f16 B, laid out exactly as the LDS tile images:
//   Bp[jt(32)][kt(32)][blk(5)][nn(16)][kk(32)]
// where logical B[k][col] (k = kt*32+kk, col = blk*512 + jt*16 + nn) is
//   col <1536 : U_iou[(k<512?0:1)][k&511][col]
//   col <2048 : k<512  ? Uf[0][k][col-1536]      : 0
//   else      : k>=512 ? Uf[1][k-512][col-2048]  : 0
// So each (jt,kt) staging step is one contiguous 5KB chunk.
// ---------------------------------------------------------------------------
__global__ __launch_bounds__(256) void prep_B(const float* __restrict__ U_iou,
                                              const float* __restrict__ Uf,
                                              _Float16* __restrict__ Bp) {
  int idx = blockIdx.x * 256 + threadIdx.x;          // < 1024*2560
  int kk = idx & 31;
  int nn = (idx >> 5) & 15;
  int chunk = idx >> 9;                               // (jt*32+kt)*5 + blk
  int blk = chunk % 5;
  int q = chunk / 5;
  int kt = q & 31;
  int jt = q >> 5;
  int k = kt * 32 + kk;
  int col = blk * 512 + jt * 16 + nn;
  float v = 0.0f;
  if (col < 1536) {
    int ch = k >> 9, kr = k & 511;                    // U_iou: (2,512,1536)
    v = U_iou[((size_t)ch * 512 + kr) * 1536 + col];
  } else if (col < 2048) {
    if (k < 512) v = Uf[(size_t)k * 512 + (col - 1536)];                  // Uf[0]
  } else {
    if (k >= 512) v = Uf[(size_t)(512 + (k - 512)) * 512 + (col - 2048)]; // Uf[1]
  }
  Bp[idx] = (_Float16)v;
}

// ---------------------------------------------------------------------------
// Leaf level (lvl=14, n=16384): hk=ck=0 -> gather + elementwise only.
// ---------------------------------------------------------------------------
__global__ __launch_bounds__(256) void leaf_kernel(const int* __restrict__ tokens,
                                                   const float* __restrict__ W_iou,
                                                   const float* __restrict__ b_iou,
                                                   _Float16* __restrict__ Hout,
                                                   float* __restrict__ Cout) {
  int idx = blockIdx.x * 256 + threadIdx.x;          // < 16384*512
  int node = idx >> 9;
  int j = idx & 511;
  int tok = tokens[16383 + node];
  const float* Wr = W_iou + (size_t)tok * 1536;
  float i = Wr[j] + b_iou[j];
  float o = Wr[512 + j] + b_iou[512 + j];
  float u = Wr[1024 + j] + b_iou[1024 + j];
  float c = sigm(i) * tanhf(u);
  float h = sigm(o) * tanhf(c);
  Cout[idx] = c;
  Hout[idx] = (_Float16)h;
}

// ---------------------------------------------------------------------------
// Internal level GEMM + fused TreeLSTM epilogue, async double-buffered.
//   A = child h buffer viewed (n x 1024) f16 (row r = [h(2r) | h(2r+1)])
//   B = packed per-(jt,kt) tile images (see prep_B)
// Grid: (ceil(n/128), 32). Each WG: 128 rows x one 16-wide j slice over the
// five strided column blocks {i,o,u,f0,f1}, so each lane holds i,o,u,f0,f1
// for the same (row, j) in its five accumulators.
// ---------------------------------------------------------------------------
__global__ __launch_bounds__(256)
void level_gemm(const _Float16* __restrict__ Ah, const float* __restrict__ Cc,
                const _Float16* __restrict__ Bp, const int* __restrict__ tokens,
                int base, int n,
                const float* __restrict__ W_iou, const float* __restrict__ b_iou,
                const float* __restrict__ Wf, const float* __restrict__ bf,
                _Float16* __restrict__ Hout, float* __restrict__ Cout) {
  __shared__ __align__(16) _Float16 sA[2][128 * 32];   // 2 x 8 KB
  __shared__ __align__(16) _Float16 sB[2][2560];       // 2 x 5 KB tile image

  const int tid = threadIdx.x;
  const int m0 = blockIdx.x * 128;
  const int jt = blockIdx.y;
  const int j0 = jt * 16;
  const int wv = tid >> 5;
  const int lane = tid & 31;
  const int hb = (lane >= 16) ? 8 : 0;                 // accumulator row half-select (M offset)
  const int kb = (lane < 16) ? 0 : 16;                 // B-frag K half-select

  const _Float16* BpT = Bp + (size_t)jt * 32 * 2560;   // this j-tile's 32 K-step images

  v8f acc[5];
#pragma unroll
  for (int t = 0; t < 5; ++t) acc[t] = v8f{};

  const int arow = tid >> 1, aseg = tid & 1;
  const int ar = m0 + arow;

  auto stage = [&](int kt, int buf) {
    if (ar < n) {
      const _Float16* src = Ah + (size_t)ar * 1024 + kt * 32 + aseg * 16;
      _Float16* dst = &sA[buf][arow * 32 + aseg * 16];
      g2l_b128(src, dst);
      g2l_b128(src + 8, dst + 8);
    }
    const _Float16* bsrc = BpT + (size_t)kt * 2560;    // contiguous 5 KB image
    g2l_b128(bsrc + tid * 8, &sB[buf][tid * 8]);
    if (tid < 64) g2l_b128(bsrc + (256 + tid) * 8, &sB[buf][(256 + tid) * 8]);
  };

  const int mrow = wv * 16 + (lane & 15);
  const int nn = lane & 15;

  auto compute = [&](int buf) {
    union { v16h v; h8 p[2]; } ua;
    ua.p[0] = *(const h8*)&sA[buf][mrow * 32 + kb];       // K = kb..kb+7
    ua.p[1] = *(const h8*)&sA[buf][mrow * 32 + kb + 8];   // K = kb+8..kb+15
    v16h bfrag[5];
#pragma unroll
    for (int t = 0; t < 5; ++t)
      bfrag[t] = *(const v16h*)&sB[buf][t * 512 + nn * 32 + kb];
#pragma unroll
    for (int t = 0; t < 5; ++t)
      acc[t] = __builtin_amdgcn_wmma_f32_16x16x32_f16(
          false, ua.v, false, bfrag[t], (short)0, acc[t], false, false);
  };

  stage(0, 0);
  wait_async();
  __syncthreads();
  for (int kt = 0; kt < 32; kt += 2) {
    stage(kt + 1, 1);          // kt+1 <= 31 always (kt <= 30)
    compute(0);
    wait_async();
    __syncthreads();
    if (kt + 2 < 32) stage(kt + 2, 0);
    compute(1);
    wait_async();
    __syncthreads();
  }

  // ---- fused epilogue: lane holds i,o,u,f0pre,f1pre for 8 rows at column j ----
  const int j = j0 + (lane & 15);
#pragma unroll
  for (int v = 0; v < 8; ++v) {
    int r = m0 + wv * 16 + v + hb;
    if (r < n) {
      int tok = tokens[base + r];
      const float* Wr = W_iou + (size_t)tok * 1536;
      float i = acc[0][v] + Wr[j] + b_iou[j];
      float o = acc[1][v] + Wr[512 + j] + b_iou[512 + j];
      float u = acc[2][v] + Wr[1024 + j] + b_iou[1024 + j];
      float fp = Wf[(size_t)tok * 512 + j] + bf[j];
      float f0 = sigm(acc[3][v] + fp);
      float f1 = sigm(acc[4][v] + fp);
      float c = sigm(i) * tanhf(u) + f0 * Cc[(size_t)(2 * r) * 512 + j]
                                   + f1 * Cc[(size_t)(2 * r + 1) * 512 + j];
      float h = sigm(o) * tanhf(c);
      Cout[(size_t)r * 512 + j] = c;
      Hout[(size_t)r * 512 + j] = (_Float16)h;
    }
  }
}

// ---------------------------------------------------------------------------
// Root ternary cell, stage 1: iou_t (1536) and fpre (3x512). Grid 12x256.
// ---------------------------------------------------------------------------
__global__ __launch_bounds__(256)
void root1(const _Float16* __restrict__ h_l0, const _Float16* __restrict__ h_l1,
           const int* __restrict__ tokens,
           const float* __restrict__ W_iou_t, const float* __restrict__ b_iou_t,
           const float* __restrict__ U_iou_t, const float* __restrict__ Wf_t,
           const float* __restrict__ bf_t, const float* __restrict__ Uf_t,
           float* __restrict__ iou_out, float* __restrict__ fpre_out) {
  __shared__ float sH[3 * 512];
  int tid = threadIdx.x;
  for (int i = tid; i < 512; i += 256) {
    sH[i] = (float)h_l0[i];
    sH[512 + i] = (float)h_l1[i];
    sH[1024 + i] = (float)h_l1[512 + i];
  }
  __syncthreads();
  int gid = blockIdx.x * 256 + tid;                     // < 3072
  int t0 = tokens[0];
  if (gid < 1536) {
    float s = b_iou_t[gid] + W_iou_t[(size_t)t0 * 1536 + gid];
#pragma unroll 1
    for (int k = 0; k < 3; ++k) {
      const float* U = U_iou_t + (size_t)k * 512 * 1536;
      float a = 0.0f;
      for (int hh = 0; hh < 512; ++hh) a += sH[k * 512 + hh] * U[(size_t)hh * 1536 + gid];
      s += a;
    }
    iou_out[gid] = s;
  } else {
    int q = gid - 1536;
    int k = q >> 9, j = q & 511;
    float s = Wf_t[(size_t)t0 * 512 + j] + bf_t[j];
    const float* U = Uf_t + (size_t)k * 512 * 512;
    for (int hh = 0; hh < 512; ++hh) s += sH[k * 512 + hh] * U[(size_t)hh * 512 + j];
    fpre_out[q] = s;
  }
}

// ---------------------------------------------------------------------------
// Root stage 2: TreeLSTM epilogue -> h_root, then FFN -> h_f. One WG of 512.
// ---------------------------------------------------------------------------
__global__ __launch_bounds__(512)
void root2(const float* __restrict__ iou_t, const float* __restrict__ fpre,
           const float* __restrict__ c_l0, const float* __restrict__ c_l1,
           const float* __restrict__ ffn_W, const float* __restrict__ ffn_b,
           float* __restrict__ hf_out) {
  __shared__ float sHr[512];
  int j = threadIdx.x;
  {
    float i = iou_t[j], o = iou_t[512 + j], u = iou_t[1024 + j];
    float c = sigm(i) * tanhf(u)
            + sigm(fpre[j]) * c_l0[j]
            + sigm(fpre[512 + j]) * c_l1[j]
            + sigm(fpre[1024 + j]) * c_l1[512 + j];
    sHr[j] = sigm(o) * tanhf(c);
  }
  __syncthreads();
  float s = ffn_b[j];
  for (int hh = 0; hh < 512; ++hh) s += sHr[hh] * ffn_W[(size_t)hh * 512 + j];
  hf_out[j] = fmaxf(s, 0.0f);
}

// ---------------------------------------------------------------------------
// LSTM gates = h_f @ lstm_Wx + lstm_b  (h_g starts at 0, so lstm_Wh is dead).
// ---------------------------------------------------------------------------
__global__ __launch_bounds__(256)
void gates_kernel(const float* __restrict__ hf, const float* __restrict__ lstm_Wx,
                  const float* __restrict__ lstm_b, float* __restrict__ gates) {
  __shared__ float sHf[512];
  int tid = threadIdx.x;
  for (int i = tid; i < 512; i += 256) sHf[i] = hf[i];
  __syncthreads();
  int g = blockIdx.x * 256 + tid;                       // < 2048
  float s = lstm_b[g];
  for (int hh = 0; hh < 512; ++hh) s += sHf[hh] * lstm_Wx[(size_t)hh * 2048 + g];
  gates[g] = s;
}

// ---------------------------------------------------------------------------
// LSTM elementwise -> feat = [h_f | h_g]; heads -> out (2 + 5 + 36 = 43).
// ---------------------------------------------------------------------------
__global__ __launch_bounds__(256)
void final_kernel(const float* __restrict__ hf, const float* __restrict__ gates,
                  const float* __restrict__ hl_W, const float* __restrict__ hl_b,
                  const float* __restrict__ int_W, const float* __restrict__ int_b,
                  const float* __restrict__ act_W, const float* __restrict__ act_b,
                  const float* __restrict__ vm, float* __restrict__ out) {
  __shared__ float feat[1024];
  int tid = threadIdx.x;
  for (int j = tid; j < 512; j += 256) {
    float gi = gates[j], gg = gates[1024 + j], go = gates[1536 + j];
    float cg = sigm(gi) * tanhf(gg);                    // c_g_prev == 0
    feat[j] = hf[j];
    feat[512 + j] = sigm(go) * tanhf(cg);
  }
  __syncthreads();
  if (tid < 43) {
    float s;
    if (tid < 2) {
      s = hl_b[tid];
      for (int hh = 0; hh < 1024; ++hh) s += feat[hh] * hl_W[hh * 2 + tid];
    } else if (tid < 7) {
      int j = tid - 2;
      s = int_b[j];
      for (int hh = 0; hh < 1024; ++hh) s += feat[hh] * int_W[hh * 5 + j];
    } else {
      int j = tid - 7;
      float vmj = vm[j];
      s = __logf(vmj) + act_b[j] * vmj;
      for (int hh = 0; hh < 1024; ++hh) s += feat[hh] * act_W[hh * 36 + j] * vmj;
    }
    out[tid] = s;
  }
}

// ---------------------------------------------------------------------------
extern "C" void kernel_launch(void* const* d_in, const int* in_sizes, int n_in,
                              void* d_out, int out_size, void* d_ws, size_t ws_size,
                              hipStream_t stream) {
  (void)in_sizes; (void)n_in; (void)out_size; (void)ws_size;
  const int*   tokens  = (const int*)d_in[0];
  const float* vm      = (const float*)d_in[1];
  const float* W_iou_b = (const float*)d_in[2];
  const float* U_iou_b = (const float*)d_in[3];
  const float* b_iou_b = (const float*)d_in[4];
  const float* Wf_b    = (const float*)d_in[5];
  const float* Uf_b    = (const float*)d_in[6];
  const float* bf_b    = (const float*)d_in[7];
  const float* W_iou_t = (const float*)d_in[8];
  const float* U_iou_t = (const float*)d_in[9];
  const float* b_iou_t = (const float*)d_in[10];
  const float* Wf_t    = (const float*)d_in[11];
  const float* Uf_t    = (const float*)d_in[12];
  const float* bf_t    = (const float*)d_in[13];
  const float* ffn_W   = (const float*)d_in[14];
  const float* ffn_b   = (const float*)d_in[15];
  const float* lstm_Wx = (const float*)d_in[16];
  /* d_in[17] = lstm_Wh unused: initial h_g == 0 */
  const float* lstm_b  = (const float*)d_in[18];
  const float* hl_W    = (const float*)d_in[19];
  const float* hl_b    = (const float*)d_in[20];
  const float* int_W   = (const float*)d_in[21];
  const float* int_b   = (const float*)d_in[22];
  const float* act_W   = (const float*)d_in[23];
  const float* act_b   = (const float*)d_in[24];

  char* ws = (char*)d_ws;
  size_t off = 0;
  auto carve = [&](size_t bytes) -> char* {
    char* p = ws + off;
    off += (bytes + 255) & ~size_t(255);
    return p;
  };
  _Float16* hA0 = (_Float16*)carve((size_t)16384 * 512 * 2);   // even levels' h
  _Float16* hA1 = (_Float16*)carve((size_t)8192 * 512 * 2);    // odd levels' h
  float*    cA0 = (float*)carve((size_t)16384 * 512 * 4);      // even levels' c
  float*    cA1 = (float*)carve((size_t)8192 * 512 * 4);       // odd levels' c
  _Float16* Bp  = (_Float16*)carve((size_t)1024 * 2560 * 2);   // packed f16 B (tile images)
  float* iou_t  = (float*)carve(1536 * 4);
  float* fpre   = (float*)carve(1536 * 4);
  float* hf     = (float*)carve(512 * 4);
  float* gbuf   = (float*)carve(2048 * 4);

  prep_B<<<(1024 * 2560) / 256, 256, 0, stream>>>(U_iou_b, Uf_b, Bp);
  leaf_kernel<<<(16384 * 512) / 256, 256, 0, stream>>>(tokens, W_iou_b, b_iou_b, hA0, cA0);

  for (int lvl = 13; lvl >= 0; --lvl) {
    int n = 1 << lvl;
    // child level = lvl+1: even child -> buf0, odd child -> buf1
    const _Float16* Ah = (lvl & 1) ? hA0 : hA1;
    const float*    Cc = (lvl & 1) ? cA0 : cA1;
    _Float16* Hout = (lvl & 1) ? hA1 : hA0;
    float*    Cout = (lvl & 1) ? cA1 : cA0;
    dim3 grid((n + 127) / 128, 32);
    level_gemm<<<grid, 256, 0, stream>>>(Ah, Cc, Bp, tokens, n - 1, n,
                                         W_iou_b, b_iou_b, Wf_b, bf_b, Hout, Cout);
  }
  // level-0 result: hA0/cA0 row 0; level-1 (h_ab/c_ab): hA1/cA1 rows 0..1
  root1<<<12, 256, 0, stream>>>(hA0, hA1, tokens, W_iou_t, b_iou_t, U_iou_t,
                                Wf_t, bf_t, Uf_t, iou_t, fpre);
  root2<<<1, 512, 0, stream>>>(iou_t, fpre, cA0, cA1, ffn_W, ffn_b, hf);
  gates_kernel<<<8, 256, 0, stream>>>(hf, lstm_Wx, lstm_b, gbuf);
  final_kernel<<<1, 256, 0, stream>>>(hf, gbuf, hl_W, hl_b, int_W, int_b,
                                      act_W, act_b, vm, (float*)d_out);
}
